// HyperNeuronDecoder_25915832664665
// MI455X (gfx1250) — matrix operand order repaired and checked by reading, not compiled, not run on hardware
//
#include <hip/hip_runtime.h>
#include <hip/hip_bf16.h>
#include <math.h>

typedef __attribute__((ext_vector_type(16))) _Float16 v16h;
typedef __attribute__((ext_vector_type(8)))  float    v8f;

#define BB   4
#define TT   128
#define RR   64
#define DS   128
#define NN   2048
#define DID  128
#define HID  256
#define OUTC 129          // Ds + 1
#define NROWS (BB*NN)     // 8192 hypernet rows

union AFrag { v16h v; _Float16 h[16]; };
union DFrag { v8f  v; float    f[8];  };

__device__ __forceinline__ float gelu_exact(float x) {
    return 0.5f * x * (1.0f + erff(x * 0.70710678118654752f));
}

// ---------------------------------------------------------------------------
// Kernel 1: group neurons by local region (counting sort), one block per b.
// ---------------------------------------------------------------------------
__global__ __launch_bounds__(256)
void group_kernel(const int* __restrict__ neuron_regions,  // [B,N]
                  const int* __restrict__ r_map,           // [128]
                  int* __restrict__ list,                  // [B,N]
                  int* __restrict__ offs,                  // [B,R]
                  int* __restrict__ cnts)                  // [B,R]
{
    __shared__ int cnt[RR];
    __shared__ int base[RR];
    __shared__ int cursor[RR];
    const int b   = blockIdx.x;
    const int tid = threadIdx.x;

    for (int r = tid; r < RR; r += blockDim.x) cnt[r] = 0;
    __syncthreads();

    for (int n = tid; n < NN; n += blockDim.x) {
        int rr = neuron_regions[b*NN + n];
        rr = rr < 0 ? 0 : (rr > 127 ? 127 : rr);
        int lr = r_map[rr];
        if (lr >= 0) {
            lr = lr < RR ? lr : (RR - 1);
            atomicAdd(&cnt[lr], 1);
        }
    }
    __syncthreads();

    if (tid == 0) {
        int acc = 0;
        for (int r = 0; r < RR; ++r) { base[r] = acc; acc += cnt[r]; }
    }
    __syncthreads();
    for (int r = tid; r < RR; r += blockDim.x) cursor[r] = base[r];
    __syncthreads();

    for (int n = tid; n < NN; n += blockDim.x) {
        int rr = neuron_regions[b*NN + n];
        rr = rr < 0 ? 0 : (rr > 127 ? 127 : rr);
        int lr = r_map[rr];
        if (lr >= 0) {
            lr = lr < RR ? lr : (RR - 1);
            int pos = atomicAdd(&cursor[lr], 1);
            list[b*NN + pos] = n;
        }
    }
    __syncthreads();
    for (int r = tid; r < RR; r += blockDim.x) {
        offs[b*RR + r] = base[r];
        cnts[b*RR + r] = cnt[r];
    }
}

// ---------------------------------------------------------------------------
// Kernel 2: hypernet.  64 rows per block, 256 threads = 8 waves (wave32).
//   phase A: embed + LayerNorm -> LDS hA (f16)
//   phase B: hA @ W1 (+b1, GELU) via v_wmma_f32_16x16x32_f16 -> LDS h1 (f16)
//   phase C: h1 @ W2 (+b2) -> w (f16, global ws) and bias (f32, global ws)
// ---------------------------------------------------------------------------
#define ROWS_PER_BLOCK 64

__global__ __launch_bounds__(256)
void hypernet_kernel(const int*   __restrict__ neuron_regions,
                     const int*   __restrict__ eids,
                     const float* __restrict__ neuron_slot,   // [2048,128]
                     const float* __restrict__ region_emb,    // [128,128]
                     const float* __restrict__ eid_emb,       // [256,128]
                     const float* __restrict__ ln_gamma,
                     const float* __restrict__ ln_beta,
                     const float* __restrict__ W1,            // [128,256]
                     const float* __restrict__ b1,            // [256]
                     const float* __restrict__ W2,            // [256,129]
                     const float* __restrict__ b2,            // [129]
                     _Float16*    __restrict__ wOut,          // [B*N,128] f16
                     float*       __restrict__ biasOut)       // [B*N]
{
    __shared__ _Float16 hA[ROWS_PER_BLOCK * DID];   // 16 KB
    __shared__ _Float16 h1[ROWS_PER_BLOCK * HID];   // 32 KB

    const int tid  = threadIdx.x;
    const int row0 = blockIdx.x * ROWS_PER_BLOCK;

    // ---- phase A: 4 threads per row, 32 dims each -------------------------
    {
        const int rl   = tid >> 2;          // local row 0..63
        const int part = tid & 3;
        const int grow = row0 + rl;
        const int b    = grow / NN;
        const int n    = grow % NN;
        int rr = neuron_regions[b*NN + n];
        rr = rr < 0 ? 0 : (rr > 127 ? 127 : rr);
        int eid = eids[b];
        eid = eid < 0 ? 0 : (eid > 255 ? 255 : eid);

        const float* ns = neuron_slot + n   * DID;
        const float* re = region_emb  + rr  * DID;
        const float* ee = eid_emb     + eid * DID;

        float s1 = 0.f, s2 = 0.f;
        const int d0 = part * 32;
        for (int d = d0; d < d0 + 32; ++d) {
            float e = ns[d] + re[d] + ee[d];
            s1 += e; s2 += e * e;
        }
        s1 += __shfl_xor(s1, 1); s1 += __shfl_xor(s1, 2);
        s2 += __shfl_xor(s2, 1); s2 += __shfl_xor(s2, 2);
        const float mu   = s1 * (1.0f / DID);
        const float var  = s2 * (1.0f / DID) - mu * mu;
        const float rstd = rsqrtf(var + 1e-5f);
        for (int d = d0; d < d0 + 32; ++d) {
            float e = ns[d] + re[d] + ee[d];
            float h = (e - mu) * rstd * ln_gamma[d] + ln_beta[d];
            hA[rl * DID + d] = (_Float16)h;
        }
    }
    __syncthreads();

    const int ww = tid >> 5;     // wave id 0..7
    const int L  = tid & 31;     // lane
    const int lm = L & 15;
    const int lh = L >> 4;

    // ---- phase B: h @ W1, GELU -> h1 --------------------------------------
    // tiles: 4 M-tiles x 16 N-tiles = 64, 8 per wave
    for (int tile = ww; tile < 4 * 16; tile += 8) {
        const int mt = tile >> 4;
        const int nt = tile & 15;
        v8f acc = {};
        #pragma unroll
        for (int ks = 0; ks < 4; ++ks) {
            AFrag a, bfr;
            #pragma unroll
            for (int i = 0; i < 16; ++i) {
                int kA = ks*32 + (i >> 3)*16 + lh*8 + (i & 7);
                a.h[i] = hA[(mt*16 + lm) * DID + kA];
            }
            #pragma unroll
            for (int i = 0; i < 16; ++i) {
                int kB = ks*32 + (i >> 3)*16 + lh*8 + (i & 7);
                bfr.h[i] = (_Float16)W1[kB * HID + (nt*16 + lm)];
            }
            acc = __builtin_amdgcn_wmma_f32_16x16x32_f16(
                false, a.v, false, bfr.v, (short)0, acc, false, false);
        }
        DFrag d; d.v = acc;
        const int ncol = nt*16 + lm;
        const float bb1 = b1[ncol];
        #pragma unroll
        for (int j = 0; j < 8; ++j) {
            const int mrow = mt*16 + j + 8*lh;
            h1[mrow * HID + ncol] = (_Float16)gelu_exact(d.f[j] + bb1);
        }
    }
    __syncthreads();

    // ---- phase C: h1 @ W2 -> w, bias --------------------------------------
    // tiles: 4 M-tiles x 9 N-tiles (144 cols, 129 valid) = 36
    for (int tile = ww; tile < 4 * 9; tile += 8) {
        const int mt = tile / 9;
        const int nt = tile % 9;
        v8f acc = {};
        #pragma unroll
        for (int ks = 0; ks < 8; ++ks) {
            AFrag a, bfr;
            #pragma unroll
            for (int i = 0; i < 16; ++i) {
                int kA = ks*32 + (i >> 3)*16 + lh*8 + (i & 7);
                a.h[i] = h1[(mt*16 + lm) * HID + kA];
            }
            #pragma unroll
            for (int i = 0; i < 16; ++i) {
                int kB = ks*32 + (i >> 3)*16 + lh*8 + (i & 7);
                int nc = nt*16 + lm;
                bfr.h[i] = (nc < OUTC) ? (_Float16)W2[kB * OUTC + nc]
                                       : (_Float16)0.0f;
            }
            acc = __builtin_amdgcn_wmma_f32_16x16x32_f16(
                false, a.v, false, bfr.v, (short)0, acc, false, false);
        }
        DFrag d; d.v = acc;
        const int ncol = nt*16 + lm;
        if (ncol < OUTC) {
            const float bb2 = b2[ncol];
            #pragma unroll
            for (int j = 0; j < 8; ++j) {
                const int grow = row0 + mt*16 + j + 8*lh;
                float v = d.f[j] + bb2;
                if (ncol < DS) wOut[grow * DS + ncol] = (_Float16)v;
                else           biasOut[grow] = v;   // ncol == 128
            }
        }
    }
}

// ---------------------------------------------------------------------------
// Kernel 3: region-grouped einsum.
// block = (b, region r, 16-neuron chunk); 8 waves cover the 8 t-tiles (T=128)
//   D[neuron, t] = sum_d w[neuron,d] * U[b,t,r,d]   (WMMA, K=128)
// ---------------------------------------------------------------------------
#define MCHUNKS (NN / 16)   // 128 worst case

__global__ __launch_bounds__(256)
void einsum_kernel(const float*    __restrict__ U,       // [B,T,R,Ds]
                   const _Float16* __restrict__ wIn,     // [B*N,128]
                   const float*    __restrict__ biasIn,  // [B*N]
                   const int*      __restrict__ list,    // [B,N]
                   const int*      __restrict__ offs,    // [B,R]
                   const int*      __restrict__ cnts,    // [B,R]
                   float*          __restrict__ out)     // [B,T,N]
{
    const int bid    = blockIdx.x;
    const int mchunk = bid % MCHUNKS;
    const int r      = (bid / MCHUNKS) % RR;
    const int b      = bid / (MCHUNKS * RR);

    const int cnt = cnts[b*RR + r];
    if (mchunk * 16 >= cnt) return;
    const int off = offs[b*RR + r];

    const int tid = threadIdx.x;
    const int ww  = tid >> 5;
    const int L   = tid & 31;
    const int lm  = L & 15;
    const int lh  = L >> 4;

    // A operand: row m = lane%16 -> neuron (same for every wave / t-tile)
    const int mrowA   = mchunk*16 + lm;
    const bool validA = (mrowA < cnt);
    const int rowA    = validA ? (b*NN + list[b*NN + off + mrowA]) : 0;

    const int t = ww*16 + lm;                 // this lane's output column
    const float* Urow = U + (((size_t)(b*TT + t)) * RR + r) * DS;

    v8f acc = {};
    #pragma unroll
    for (int ks = 0; ks < 4; ++ks) {
        AFrag a, bfr;
        #pragma unroll
        for (int i = 0; i < 16; ++i) {
            int kA = ks*32 + (i >> 3)*16 + lh*8 + (i & 7);
            a.h[i] = validA ? wIn[rowA * DS + kA] : (_Float16)0.0f;
        }
        #pragma unroll
        for (int i = 0; i < 16; ++i) {
            int d = ks*32 + (i >> 3)*16 + lh*8 + (i & 7);
            bfr.h[i] = (_Float16)Urow[d];
        }
        acc = __builtin_amdgcn_wmma_f32_16x16x32_f16(
            false, a.v, false, bfr.v, (short)0, acc, false, false);
    }

    DFrag d; d.v = acc;
    #pragma unroll
    for (int j = 0; j < 8; ++j) {
        const int ml = mchunk*16 + j + 8*lh;
        if (ml < cnt) {
            const int n = list[b*NN + off + ml];
            out[((size_t)(b*TT + t)) * NN + n] = d.f[j] + biasIn[b*NN + n];
        }
    }
}

// ---------------------------------------------------------------------------
// Launcher
// ---------------------------------------------------------------------------
extern "C" void kernel_launch(void* const* d_in, const int* in_sizes, int n_in,
                              void* d_out, int out_size, void* d_ws, size_t ws_size,
                              hipStream_t stream) {
    const float* U              = (const float*)d_in[0];
    const int*   neuron_regions = (const int*)  d_in[1];
    const int*   eids           = (const int*)  d_in[2];
    const int*   r_map          = (const int*)  d_in[3];
    const float* neuron_slot    = (const float*)d_in[4];
    const float* region_emb     = (const float*)d_in[5];
    const float* eid_emb        = (const float*)d_in[6];
    const float* ln_gamma       = (const float*)d_in[7];
    const float* ln_beta        = (const float*)d_in[8];
    const float* W1             = (const float*)d_in[9];
    const float* b1             = (const float*)d_in[10];
    const float* W2             = (const float*)d_in[11];
    const float* b2             = (const float*)d_in[12];
    float* out = (float*)d_out;

    // workspace layout (all chunks are 1 KB multiples -> aligned)
    char* ws = (char*)d_ws;
    size_t o = 0;
    _Float16* wBuf    = (_Float16*)(ws + o); o += (size_t)NROWS * DS * sizeof(_Float16); // 2 MB
    float*    biasBuf = (float*)   (ws + o); o += (size_t)NROWS * sizeof(float);          // 32 KB
    int*      list    = (int*)     (ws + o); o += (size_t)NROWS * sizeof(int);            // 32 KB
    int*      offs    = (int*)     (ws + o); o += (size_t)BB * RR * sizeof(int);          // 1 KB
    int*      cnts    = (int*)     (ws + o); o += (size_t)BB * RR * sizeof(int);          // 1 KB
    (void)ws_size; (void)o;

    // masked / unwritten predictions must be zero
    hipMemsetAsync(d_out, 0, (size_t)out_size * sizeof(float), stream);

    group_kernel<<<BB, 256, 0, stream>>>(neuron_regions, r_map, list, offs, cnts);

    hypernet_kernel<<<NROWS / ROWS_PER_BLOCK, 256, 0, stream>>>(
        neuron_regions, eids, neuron_slot, region_emb, eid_emb,
        ln_gamma, ln_beta, W1, b1, W2, b2, wBuf, biasBuf);

    einsum_kernel<<<BB * RR * MCHUNKS, 256, 0, stream>>>(
        U, wBuf, biasBuf, list, offs, cnts, out);
}
